// RelicLoss_31507880083507
// MI455X (gfx1250) — compile-verified, matrix operand order checked by reading, not run on hardware
//
#include <hip/hip_runtime.h>

#define BS  4096
#define DIMK 1024

typedef __attribute__((ext_vector_type(16))) __bf16 v16bf;
typedef __attribute__((ext_vector_type(8)))  __bf16 bf16x8;
typedef __attribute__((ext_vector_type(8)))  float  v8f;

union Frag { v16bf v; bf16x8 h[2]; };

__device__ __forceinline__ unsigned short f2bf(float f) {
  unsigned int u = __float_as_uint(f);
  u += 0x7FFFu + ((u >> 16) & 1u);   // round-to-nearest-even
  return (unsigned short)(u >> 16);
}

// ---------------------------------------------------------------- zeroing ---
__global__ void zero_kernel(float* __restrict__ p, int n) {
  int i = blockIdx.x * 256 + threadIdx.x;
  if (i < n) p[i] = 0.0f;
}
__global__ void zero_small_kernel(float* __restrict__ S, float* __restrict__ out) {
  if (threadIdx.x == 0) { S[0] = 0.0f; S[1] = 0.0f; out[0] = 0.0f; }
}

// ------------------------------------------- normalize + bf16 + diag dots ---
__global__ __launch_bounds__(256) void norm_kernel(
    const float* __restrict__ zi, const float* __restrict__ zj,
    const float* __restrict__ zo,
    unsigned short* __restrict__ xi, unsigned short* __restrict__ xj,
    float* __restrict__ dio, float* __restrict__ djo) {
  const int r = blockIdx.x;
  const int t = threadIdx.x;
  const float4* pi = (const float4*)(zi + (size_t)r * DIMK);
  const float4* pj = (const float4*)(zj + (size_t)r * DIMK);
  const float4* po = (const float4*)(zo + (size_t)r * DIMK);
  float4 a = pi[t], b = pj[t], c = po[t];

  float vals[5];
  vals[0] = a.x*a.x + a.y*a.y + a.z*a.z + a.w*a.w;   // ||zi||^2
  vals[1] = b.x*b.x + b.y*b.y + b.z*b.z + b.w*b.w;   // ||zj||^2
  vals[2] = c.x*c.x + c.y*c.y + c.z*c.z + c.w*c.w;   // ||zo||^2
  vals[3] = a.x*c.x + a.y*c.y + a.z*c.z + a.w*c.w;   // zi.zo
  vals[4] = b.x*c.x + b.y*c.y + b.z*c.z + b.w*c.w;   // zj.zo

  __shared__ float sh[5][8];
  __shared__ float tot[5];
  const int w = t >> 5, lane = t & 31;
  #pragma unroll
  for (int q = 0; q < 5; ++q) {
    float s = vals[q];
    #pragma unroll
    for (int off = 16; off >= 1; off >>= 1) s += __shfl_xor(s, off, 32);
    if (lane == 0) sh[q][w] = s;
  }
  __syncthreads();
  if (t < 5) {
    float s = 0.0f;
    #pragma unroll
    for (int k = 0; k < 8; ++k) s += sh[t][k];
    tot[t] = s;
  }
  __syncthreads();

  float si = tot[0], sj = tot[1], so = tot[2];
  float rni = __frsqrt_rn(si); rni = rni * (1.5f - 0.5f * si * rni * rni);
  float rnj = __frsqrt_rn(sj); rnj = rnj * (1.5f - 0.5f * sj * rnj * rnj);
  float rno = __frsqrt_rn(so); rno = rno * (1.5f - 0.5f * so * rno * rno);

  if (t == 0) {
    dio[r] = tot[3] * rni * rno;
    djo[r] = tot[4] * rnj * rno;
  }
  size_t base = (size_t)r * DIMK + (size_t)t * 4;
  xi[base + 0] = f2bf(a.x * rni); xi[base + 1] = f2bf(a.y * rni);
  xi[base + 2] = f2bf(a.z * rni); xi[base + 3] = f2bf(a.w * rni);
  xj[base + 0] = f2bf(b.x * rnj); xj[base + 1] = f2bf(b.y * rnj);
  xj[base + 2] = f2bf(b.z * rnj); xj[base + 3] = f2bf(b.w * rnj);
}

// ---------------------------------------------------- fused exp-sum GEMM ----
// L = X @ Y^T via v_wmma_f32_16x16x32_bf16 with a 2-stage software pipeline.
// Epilogue fuses exp + row/col sums + diagonal capture (matrices never stored).
// Workgroup = 8 waves, tile 128x128; wave tile 64x32 = 4x2 fragments.

__device__ __forceinline__ void load_frags(
    Frag a[4], Frag b[2],
    const unsigned short* const aptr[4], const unsigned short* const bptr[2],
    int kk) {
  #pragma unroll
  for (int fm = 0; fm < 4; ++fm) {
    const bf16x8* p = (const bf16x8*)(aptr[fm] + kk);
    a[fm].h[0] = p[0];          // K = khalf+0..7
    a[fm].h[1] = p[2];          // K = khalf+16..23
  }
  #pragma unroll
  for (int fn = 0; fn < 2; ++fn) {
    const bf16x8* p = (const bf16x8*)(bptr[fn] + kk);
    b[fn].h[0] = p[0];
    b[fn].h[1] = p[2];
  }
}

__device__ __forceinline__ void do_wmma(v8f acc[4][2], const Frag a[4],
                                        const Frag b[2]) {
  #pragma unroll
  for (int fm = 0; fm < 4; ++fm)
    #pragma unroll
    for (int fn = 0; fn < 2; ++fn)
      acc[fm][fn] = __builtin_amdgcn_wmma_f32_16x16x32_bf16(
          false, a[fm].v, false, b[fn].v, (short)0, acc[fm][fn], false, false);
}

__global__ __launch_bounds__(256) void simexp_gemm(
    const unsigned short* __restrict__ X, const unsigned short* __restrict__ Y,
    float* __restrict__ rowsum, float* __restrict__ colsum,
    float* __restrict__ diag) {
  const int lane  = threadIdx.x & 31;
  const int w     = threadIdx.x >> 5;
  const int lr    = lane & 15;
  const int hi    = lane >> 4;
  const int khalf = hi * 8;                 // K-half split per ISA A/B layout
  const int mo = (w & 1) * 64;
  const int no = (w >> 1) * 32;
  const int rowBase = blockIdx.y * 128 + mo;
  const int colBase = blockIdx.x * 128 + no;

  const unsigned short* aptr[4];
  const unsigned short* bptr[2];
  #pragma unroll
  for (int fm = 0; fm < 4; ++fm)
    aptr[fm] = X + (size_t)(rowBase + fm * 16 + lr) * DIMK + khalf;
  #pragma unroll
  for (int fn = 0; fn < 2; ++fn)
    bptr[fn] = Y + (size_t)(colBase + fn * 16 + lr) * DIMK + khalf;

  const v8f vzero = {0.f, 0.f, 0.f, 0.f, 0.f, 0.f, 0.f, 0.f};
  v8f acc[4][2];
  #pragma unroll
  for (int fm = 0; fm < 4; ++fm)
    #pragma unroll
    for (int fn = 0; fn < 2; ++fn) acc[fm][fn] = vzero;

  // 2-stage software pipeline, ping-pong fragment buffers, rolled loop:
  // each WMMA group depends only on loads issued one stage earlier.
  Frag a0[4], b0[2], a1[4], b1[2];
  load_frags(a0, b0, aptr, bptr, 0);
  #pragma unroll 1
  for (int kk = 0; kk < DIMK; kk += 64) {
    load_frags(a1, b1, aptr, bptr, kk + 32);
    do_wmma(acc, a0, b0);
    const int kn = (kk + 64 < DIMK) ? kk + 64 : 0;   // safe dummy prefetch
    load_frags(a0, b0, aptr, bptr, kn);
    do_wmma(acc, a1, b1);
  }

  // Epilogue: C/D layout -> element (vgpr v, lane): M = v + 8*hi, N = lr.
  float cs0 = 0.0f, cs1 = 0.0f;
  const int gn0 = colBase + lr;
  const int gn1 = colBase + 16 + lr;
  #pragma unroll
  for (int fm = 0; fm < 4; ++fm) {
    #pragma unroll
    for (int v = 0; v < 8; ++v) {
      const int gm = rowBase + fm * 16 + v + 8 * hi;
      float r0 = acc[fm][0][v];
      float r1 = acc[fm][1][v];
      if (diag) {                       // raw logit on the diagonal
        if (gm == gn0) diag[gm] = r0;
        if (gm == gn1) diag[gm] = r1;
      }
      float e0 = __expf(r0);
      float e1 = __expf(r1);
      cs0 += e0; cs1 += e1;
      float s = e0 + e1;
      #pragma unroll
      for (int off = 8; off >= 1; off >>= 1) s += __shfl_xor(s, off, 32);
      if (lr == 0) atomicAdd(&rowsum[gm], s);   // lanes 0/16 -> M=v, v+8
    }
  }
  if (colsum) {
    cs0 += __shfl_xor(cs0, 16, 32);
    cs1 += __shfl_xor(cs1, 16, 32);
    if (hi == 0) {
      atomicAdd(&colsum[gn0], cs0);
      atomicAdd(&colsum[gn1], cs1);
    }
  }
}

// ------------------------------------------------ KL softmax denominators ---
__global__ __launch_bounds__(256) void klsum_kernel(
    const float* __restrict__ dio, const float* __restrict__ djo,
    float* __restrict__ S) {
  int i = blockIdx.x * 256 + threadIdx.x;
  float ei = __expf(dio[i]);
  float ej = __expf(djo[i]);
  __shared__ float sh[2][8];
  int w = threadIdx.x >> 5, lane = threadIdx.x & 31;
  #pragma unroll
  for (int off = 16; off >= 1; off >>= 1) {
    ei += __shfl_xor(ei, off, 32);
    ej += __shfl_xor(ej, off, 32);
  }
  if (lane == 0) { sh[0][w] = ei; sh[1][w] = ej; }
  __syncthreads();
  if (threadIdx.x == 0) {
    float a = 0.f, b = 0.f;
    #pragma unroll
    for (int k = 0; k < 8; ++k) { a += sh[0][k]; b += sh[1][k]; }
    atomicAdd(&S[0], a);
    atomicAdd(&S[1], b);
  }
}

// ----------------------------------------------------------------- final ----
__global__ __launch_bounds__(256) void finalize_kernel(
    const float* __restrict__ rs_ii, const float* __restrict__ rs_ij,
    const float* __restrict__ cs_ij, const float* __restrict__ rs_jj,
    const float* __restrict__ d_ii, const float* __restrict__ d_ij,
    const float* __restrict__ d_jj,
    const float* __restrict__ dio, const float* __restrict__ djo,
    const float* __restrict__ S, float* __restrict__ out) {
  int i = blockIdx.x * 256 + threadIdx.x;
  float lse1 = logf(rs_ii[i] - __expf(d_ii[i]) + rs_ij[i]);
  float lse2 = logf(cs_ij[i] + rs_jj[i] - __expf(d_jj[i]));
  float contrib = (lse1 + lse2 - 2.0f * d_ij[i]) * (1.0f / (2.0f * BS));
  float lsm = djo[i] - logf(S[1]);
  float p   = __expf(lsm);
  float sm  = __expf(dio[i]) / S[0];
  float val = contrib + 0.5f * (p * (lsm - sm));

  __shared__ float sh[8];
  int w = threadIdx.x >> 5, lane = threadIdx.x & 31;
  #pragma unroll
  for (int off = 16; off >= 1; off >>= 1) val += __shfl_xor(val, off, 32);
  if (lane == 0) sh[w] = val;
  __syncthreads();
  if (threadIdx.x == 0) {
    float a = 0.f;
    #pragma unroll
    for (int k = 0; k < 8; ++k) a += sh[k];
    atomicAdd(out, a);
  }
}

// ---------------------------------------------------------------- launch ----
extern "C" void kernel_launch(void* const* d_in, const int* in_sizes, int n_in,
                              void* d_out, int out_size, void* d_ws, size_t ws_size,
                              hipStream_t stream) {
  (void)in_sizes; (void)n_in; (void)out_size; (void)ws_size;
  const float* zi = (const float*)d_in[0];
  const float* zj = (const float*)d_in[1];
  const float* zo = (const float*)d_in[2];
  float* out = (float*)d_out;

  char* ws = (char*)d_ws;
  unsigned short* Xi = (unsigned short*)ws;                      // 8 MB bf16
  unsigned short* Xj = Xi + (size_t)BS * DIMK;                   // 8 MB bf16
  float* fbase = (float*)(ws + (size_t)2 * BS * DIMK * sizeof(unsigned short));
  float* rs_ii = fbase + 0 * BS;
  float* rs_ij = fbase + 1 * BS;
  float* cs_ij = fbase + 2 * BS;
  float* rs_jj = fbase + 3 * BS;
  float* d_ii  = fbase + 4 * BS;
  float* d_ij  = fbase + 5 * BS;
  float* d_jj  = fbase + 6 * BS;
  float* dio   = fbase + 7 * BS;
  float* djo   = fbase + 8 * BS;
  float* S     = fbase + 9 * BS;   // S[0]=sum exp(dio), S[1]=sum exp(djo)

  zero_kernel<<<(4 * BS + 255) / 256, 256, 0, stream>>>(fbase, 4 * BS);
  zero_small_kernel<<<1, 32, 0, stream>>>(S, out);

  norm_kernel<<<BS, 256, 0, stream>>>(zi, zj, zo, Xi, Xj, dio, djo);

  dim3 grid(BS / 128, BS / 128);   // 32 x 32 tiles of 128x128
  simexp_gemm<<<grid, 256, 0, stream>>>(Xi, Xi, rs_ii, nullptr, d_ii);
  simexp_gemm<<<grid, 256, 0, stream>>>(Xi, Xj, rs_ij, cs_ij,  d_ij);
  simexp_gemm<<<grid, 256, 0, stream>>>(Xj, Xj, rs_jj, nullptr, d_jj);

  klsum_kernel<<<BS / 256, 256, 0, stream>>>(dio, djo, S);
  finalize_kernel<<<BS / 256, 256, 0, stream>>>(rs_ii, rs_ij, cs_ij, rs_jj,
                                                d_ii, d_ij, d_jj, dio, djo, S, out);
}
